// MABpp_84911503442129
// MI455X (gfx1250) — matrix operand order repaired and checked by reading, not compiled, not run on hardware
//
#include <hip/hip_runtime.h>
#include <hip/hip_bf16.h>

// MI455X / gfx1250, wave32. All matmuls via v_wmma_f32_16x16x32_f16.
// Hot loops are unrolled x2 with ping-pong register buffers: prefetch loads
// write directly into the set consumed next stage (no v_mov rotation).

typedef _Float16 f16;
typedef _Float16 v16h __attribute__((ext_vector_type(16)));
typedef _Float16 v8h  __attribute__((ext_vector_type(8)));
typedef _Float16 v4h  __attribute__((ext_vector_type(4)));
typedef float    v8f  __attribute__((ext_vector_type(8)));

#define BB    2
#define NSEQ  2048
#define CC    512
#define HH    8
#define DD    64
#define EPSN  1e-5f
#define MROWS (BB*NSEQ)   // 4096

static __device__ __forceinline__ v8f wmma16(v16h a, v16h b, v8f c) {
  // (neg_a, A, neg_b, B, c_mod, C, reuse_a, reuse_b)
  return __builtin_amdgcn_wmma_f32_16x16x32_f16(false, a, false, b, (short)0, c, false, false);
}

// A-operand (16x32 f16) per-lane load: elements 0..7 = K[base..base+7],
// elements 8..15 = K[base+16..base+23]; base includes +8*(lane/16).
static __device__ __forceinline__ v16h load_a(const f16* p) {
  v8h lo = *(const v8h*)(p);
  v8h hi = *(const v8h*)(p + 16);
  v16h r;
#pragma unroll
  for (int e = 0; e < 8; ++e) { r[e] = lo[e]; r[e + 8] = hi[e]; }
  return r;
}

// ---------------- set-norm statistics (deterministic 2-stage) ----------------

__global__ __launch_bounds__(256) void k_stats_part(const float* __restrict__ X,
                                                    float* __restrict__ part, int perBatch) {
  int b = blockIdx.y;
  const float4* Xv = (const float4*)(X + (size_t)b * perBatch);
  int n4 = perBatch >> 2;
  float s = 0.f, s2 = 0.f;
  for (int i = blockIdx.x * blockDim.x + threadIdx.x; i < n4; i += gridDim.x * blockDim.x) {
    float4 v = Xv[i];
    s  += v.x + v.y + v.z + v.w;
    s2 += v.x * v.x + v.y * v.y + v.z * v.z + v.w * v.w;
  }
#pragma unroll
  for (int off = 16; off > 0; off >>= 1) {
    s  += __shfl_down(s, off, 32);
    s2 += __shfl_down(s2, off, 32);
  }
  __shared__ float sh[16];
  int wave = threadIdx.x >> 5, lane = threadIdx.x & 31;
  if (lane == 0) { sh[wave * 2] = s; sh[wave * 2 + 1] = s2; }
  __syncthreads();
  if (threadIdx.x == 0) {
    float ts = 0.f, t2 = 0.f;
    for (int w2 = 0; w2 < 8; ++w2) { ts += sh[2 * w2]; t2 += sh[2 * w2 + 1]; }
    part[(b * 64 + blockIdx.x) * 2 + 0] = ts;
    part[(b * 64 + blockIdx.x) * 2 + 1] = t2;
  }
}

__global__ __launch_bounds__(32) void k_stats_final(const float* __restrict__ part,
                                                    float* __restrict__ out2) {
  int b = blockIdx.x;
  int t = threadIdx.x;  // 32 threads, 64 partial pairs
  const float* p = part + b * 128;
  float s  = p[2 * t + 0] + p[2 * (t + 32) + 0];
  float s2 = p[2 * t + 1] + p[2 * (t + 32) + 1];
#pragma unroll
  for (int off = 16; off > 0; off >>= 1) {
    s  += __shfl_down(s, off, 32);
    s2 += __shfl_down(s2, off, 32);
  }
  if (t == 0) { out2[2 * b] = s; out2[2 * b + 1] = s2; }
}

// ---------------- normalize (fp32 -> f16) ----------------

__global__ __launch_bounds__(256) void k_norm(const float* __restrict__ X,
                                              const float* __restrict__ st,
                                              const float* __restrict__ gamma,
                                              const float* __restrict__ beta,
                                              f16* __restrict__ Y, int perBatch, float invCount) {
  int total4 = (BB * perBatch) >> 2;
  for (int i = blockIdx.x * blockDim.x + threadIdx.x; i < total4; i += gridDim.x * blockDim.x) {
    int idx = i << 2;
    int b = idx / perBatch;
    int c = idx & (CC - 1);
    float mu  = st[2 * b] * invCount;
    float var = st[2 * b + 1] * invCount - mu * mu;
    float inv = rsqrtf(var + EPSN);
    float4 x = *(const float4*)(X + idx);
    v4h y;
    y[0] = (f16)((x.x - mu) * inv * gamma[c + 0] + beta[c + 0]);
    y[1] = (f16)((x.y - mu) * inv * gamma[c + 1] + beta[c + 1]);
    y[2] = (f16)((x.z - mu) * inv * gamma[c + 2] + beta[c + 2]);
    y[3] = (f16)((x.w - mu) * inv * gamma[c + 3] + beta[c + 3]);
    *(v4h*)(Y + idx) = y;
  }
}

// ---------------- pack fp32 weights into WMMA B-operand layout ----------------
// Wp[((nt*16 + kt)*32 + lane)*16 + e] = W[kt*32 + 8*(lane/16) + e + (e>=8 ? 8 : 0)][nt*16 + lane%16]

__global__ __launch_bounds__(256) void k_pack(const float* __restrict__ W, f16* __restrict__ Wp) {
  int idx = blockIdx.x * blockDim.x + threadIdx.x;  // 32*16*32 = 16384
  if (idx >= 32 * 16 * 32) return;
  int lane = idx & 31;
  int kt = (idx >> 5) & 15;
  int nt = idx >> 9;
  int n = nt * 16 + (lane & 15);
  int kbase = kt * 32 + ((lane >> 4) << 3);
  v16h y;
#pragma unroll
  for (int e = 0; e < 16; ++e) y[e] = (f16)W[(kbase + e + (e >= 8 ? 8 : 0)) * CC + n];
  *(v16h*)(Wp + (size_t)idx * 16) = y;
}

// ---------------- GEMM: [M,512] x [512,512] + bias, three epilogues ----------------
// epi 0: f16 row-major store; epi 1: f16 transposed store [B,H,D,NKV];
// epi 2: fp32 out = resid + relu(acc + bias)
// Ping-pong double buffering, unrolled x2: no rotation moves.

__global__ __launch_bounds__(256) void k_gemm(const f16* __restrict__ A,
                                              const f16* __restrict__ Wp,
                                              const float* __restrict__ bias,
                                              f16* __restrict__ outH,
                                              const float* __restrict__ resid,
                                              float* __restrict__ outF, int epi) {
  int lane = threadIdx.x & 31;
  int wave = threadIdx.x >> 5;
  int hi = lane >> 4;
  int rowbase = blockIdx.x * 16;
  const f16* arow = A + (size_t)(rowbase + (lane & 15)) * CC + hi * 8;
  const f16* wrow = Wp + ((size_t)(wave * 4 * 16) * 32 + lane) * 16;  // nt=wave*4, kt=0

  v8f acc[4] = {};
  // stage 0 buffers
  v16h a0 = load_a(arow);
  v16h b0[4];
#pragma unroll
  for (int j = 0; j < 4; ++j) b0[j] = *(const v16h*)(wrow + (size_t)(j * 16) * 32 * 16);

  for (int kt = 0; kt < 16; kt += 2) {
    int k1 = (kt + 1) & 15;
    int k2 = (kt + 2) & 15;   // wrapped: branch-free, last reload is harmless
    // prefetch stage 1
    v16h a1 = load_a(arow + k1 * 32);
    v16h b1[4];
#pragma unroll
    for (int j = 0; j < 4; ++j)
      b1[j] = *(const v16h*)(wrow + (size_t)(j * 16 + k1) * 32 * 16);
    // consume stage 0
#pragma unroll
    for (int j = 0; j < 4; ++j) acc[j] = wmma16(a0, b0[j], acc[j]);
    // prefetch stage 0 for next iteration (loads write phi regs directly)
    a0 = load_a(arow + k2 * 32);
#pragma unroll
    for (int j = 0; j < 4; ++j)
      b0[j] = *(const v16h*)(wrow + (size_t)(j * 16 + k2) * 32 * 16);
    // consume stage 1
#pragma unroll
    for (int j = 0; j < 4; ++j) acc[j] = wmma16(a1, b1[j], acc[j]);
  }

#pragma unroll
  for (int j = 0; j < 4; ++j) {
    int nt = wave * 4 + j;
    int n = nt * 16 + (lane & 15);
    float bn = bias[n];
    if (epi == 0) {
#pragma unroll
      for (int r = 0; r < 8; ++r) {
        int m = rowbase + hi * 8 + r;
        outH[(size_t)m * CC + n] = (f16)(acc[j][r] + bn);
      }
    } else if (epi == 1) {
      int m0 = rowbase + hi * 8;
      int bb = m0 >> 11;          // / NSEQ
      int kv0 = m0 & (NSEQ - 1);
      int h = n >> 6, d = n & 63;
      v8h pk;
#pragma unroll
      for (int r = 0; r < 8; ++r) pk[r] = (f16)(acc[j][r] + bn);
      *(v8h*)(outH + ((size_t)((bb * HH + h) * DD + d)) * NSEQ + kv0) = pk;
    } else {
#pragma unroll
      for (int r = 0; r < 8; ++r) {
        int m = rowbase + hi * 8 + r;
        float v = acc[j][r] + bn;
        v = v > 0.f ? v : 0.f;
        size_t off = (size_t)m * CC + n;
        outF[off] = resid[off] + v;
      }
    }
  }
}

// ---------------- flash attention, one 16-query tile per wave ----------------
// Computes S^T = K . Q^T so exp(S) lands directly in the A-operand layout for P.V.
// kv loop unrolled x2 with ping-pong K-tile sets (no rotation moves); V loads
// issued first; next-chunk K loads issued behind the score WMMAs; softmax
// VALU/exp chain hides both.

struct AttnState {
  v8f o0, o1, o2, o3;
  float m_run, l_run;
};

static __device__ __forceinline__ void attn_chunk(
    AttnState& st, int kvb, int hi, int ql,
    const f16* __restrict__ kbase, const f16* __restrict__ vbase,
    v16h bq0, v16h bq1,
    v16h a00, v16h a01, v16h a10, v16h a11,          // current chunk K tiles
    v16h& n00, v16h& n01, v16h& n10, v16h& n11) {    // prefetch target (next)
  const float scale = 0.04419417382415922f;  // 1/sqrt(512)

  // V loads for this chunk (independent of scores) issued first
  const f16* vp = vbase + kvb;
  v16h v0 = load_a(vp + (size_t)(0 * 16 + ql) * NSEQ);
  v16h v1 = load_a(vp + (size_t)(1 * 16 + ql) * NSEQ);
  v16h v2 = load_a(vp + (size_t)(2 * 16 + ql) * NSEQ);
  v16h v3 = load_a(vp + (size_t)(3 * 16 + ql) * NSEQ);

  // score WMMAs from previously-loaded K tiles (interleaved accumulators)
  v8f s0 = {}, s1 = {};
  s0 = wmma16(a00, bq0, s0);
  s1 = wmma16(a10, bq0, s1);
  s0 = wmma16(a01, bq1, s0);
  s1 = wmma16(a11, bq1, s1);

  // prefetch next chunk's K tiles (wrapped address, branch-free)
  int nkv = (kvb + 32) & (NSEQ - 1);
  const f16* kp0 = kbase + (size_t)(nkv + ql) * CC;
  const f16* kp1 = kbase + (size_t)(nkv + 16 + ql) * CC;
  n00 = load_a(kp0);
  n01 = load_a(kp0 + 32);
  n10 = load_a(kp1);
  n11 = load_a(kp1 + 32);

  // online softmax (VALU/trans chain hides V + K-prefetch latency)
  float mx = -1e30f;
#pragma unroll
  for (int r = 0; r < 8; ++r) {
    s0[r] *= scale; s1[r] *= scale;
    mx = fmaxf(mx, fmaxf(s0[r], s1[r]));
  }
  mx = fmaxf(mx, __shfl_xor(mx, 16, 32));
  float newm = fmaxf(st.m_run, mx);
  float alpha = __expf(st.m_run - newm);
  st.m_run = newm;

  v16h p;
  float rs = 0.f;
#pragma unroll
  for (int r = 0; r < 8; ++r) {
    float e0 = __expf(s0[r] - newm);   // kv offset r + 8*hi      -> A elem r
    float e1 = __expf(s1[r] - newm);   // kv offset 16 + r + 8*hi -> A elem r+8
    rs += e0 + e1;
    p[r] = (f16)e0; p[r + 8] = (f16)e1;
  }
  rs += __shfl_xor(rs, 16, 32);
  st.l_run = st.l_run * alpha + rs;

  // rescale O by per-q-row alpha (O rows live at q = r + 8*hi)
#pragma unroll
  for (int r = 0; r < 8; ++r) {
    float ar = __shfl(alpha, r + hi * 8, 32);
    st.o0[r] *= ar; st.o1[r] *= ar; st.o2[r] *= ar; st.o3[r] *= ar;
  }

  // P.V WMMAs (V loads completed while softmax ran)
  st.o0 = wmma16(p, v0, st.o0);
  st.o1 = wmma16(p, v1, st.o1);
  st.o2 = wmma16(p, v2, st.o2);
  st.o3 = wmma16(p, v3, st.o3);
}

__global__ __launch_bounds__(256) void k_attn(const f16* __restrict__ Qh,
                                              const f16* __restrict__ Kh,
                                              const f16* __restrict__ VT,
                                              const float* __restrict__ query,
                                              float* __restrict__ O1) {
  int lane = threadIdx.x & 31;
  int wave = threadIdx.x >> 5;
  int gw = blockIdx.x * (blockDim.x >> 5) + wave;   // 0..2047
  int qt = gw & 127;
  int h  = (gw >> 7) & 7;
  int b  = gw >> 10;
  int hi = lane >> 4;
  int ql = lane & 15;
  int qbase = qt * 16;

  // Q as B-operand: per lane q = ql fixed, d at 8*hi..+7 and 16+8*hi..+7 (+32 per k-step)
  const f16* qp = Qh + ((size_t)(b * NSEQ + qbase + ql)) * CC + h * DD + hi * 8;
  v16h bq0 = load_a(qp);
  v16h bq1 = load_a(qp + 32);

  AttnState st;
  st.o0 = v8f{}; st.o1 = v8f{}; st.o2 = v8f{}; st.o3 = v8f{};
  st.m_run = -1e30f; st.l_run = 0.f;

  const f16* kbase = Kh + ((size_t)b * NSEQ) * CC + h * DD + hi * 8;
  const f16* vbase = VT + ((size_t)(b * HH + h)) * DD * NSEQ + hi * 8;

  // prologue: K A-tiles for chunk 0 (set X)
  v16h x00 = load_a(kbase + (size_t)(0 + ql) * CC);
  v16h x01 = load_a(kbase + (size_t)(0 + ql) * CC + 32);
  v16h x10 = load_a(kbase + (size_t)(16 + ql) * CC);
  v16h x11 = load_a(kbase + (size_t)(16 + ql) * CC + 32);
  v16h y00, y01, y10, y11;  // set Y, filled by first chunk's prefetch

  for (int kvb = 0; kvb < NSEQ; kvb += 64) {
    // chunk kvb: consume X, prefetch into Y
    attn_chunk(st, kvb, hi, ql, kbase, vbase, bq0, bq1,
               x00, x01, x10, x11, y00, y01, y10, y11);
    // chunk kvb+32: consume Y, prefetch into X
    attn_chunk(st, kvb + 32, hi, ql, kbase, vbase, bq0, bq1,
               y00, y01, y10, y11, x00, x01, x10, x11);
  }

  float linv = 1.f / st.l_run;
#pragma unroll
  for (int r = 0; r < 8; ++r) {
    float lr = __shfl(linv, r + hi * 8, 32);
    int q = qbase + hi * 8 + r;
    size_t rowoff = ((size_t)(b * NSEQ + q)) * CC + h * DD + ql;
    O1[rowoff +  0] = st.o0[r] * lr + query[rowoff +  0];
    O1[rowoff + 16] = st.o1[r] * lr + query[rowoff + 16];
    O1[rowoff + 32] = st.o2[r] * lr + query[rowoff + 32];
    O1[rowoff + 48] = st.o3[r] * lr + query[rowoff + 48];
  }
}

// ---------------- host launcher ----------------

extern "C" void kernel_launch(void* const* d_in, const int* in_sizes, int n_in,
                              void* d_out, int out_size, void* d_ws, size_t ws_size,
                              hipStream_t stream) {
  const float* query = (const float*)d_in[0];
  const float* keyv  = (const float*)d_in[1];
  const float* Wq = (const float*)d_in[2];
  const float* bq = (const float*)d_in[3];
  const float* Wk = (const float*)d_in[4];
  const float* bk = (const float*)d_in[5];
  const float* Wv = (const float*)d_in[6];
  const float* bv = (const float*)d_in[7];
  const float* Wo = (const float*)d_in[8];
  const float* bo = (const float*)d_in[9];
  const float* gq  = (const float*)d_in[10];
  const float* btq = (const float*)d_in[11];
  const float* gk  = (const float*)d_in[12];
  const float* btk = (const float*)d_in[13];
  const float* go  = (const float*)d_in[14];
  const float* bto = (const float*)d_in[15];
  float* out = (float*)d_out;

  // workspace layout (~34.3 MB)
  const size_t NH = (size_t)MROWS * CC;            // 2M halves per activation buf
  float* stats = (float*)d_ws;                     // finals [0..11]
  float* part  = stats + 16;                       // 256 partial floats
  f16* qn  = (f16*)((char*)d_ws + 8192);
  f16* kvn = qn  + NH;
  f16* Qm  = kvn + NH;
  f16* Km  = Qm  + NH;
  f16* VTt = Km  + NH;
  f16* On  = VTt + NH;
  f16* Wp  = On  + NH;                             // 4 packed 512x512 f16 weights
  float* O1 = (float*)(Wp + 4 * (size_t)CC * CC);  // 8 MB fp32 residual buffer

  const int perBatch = NSEQ * CC;
  const float invCount = 1.f / (float)perBatch;
  dim3 sgrid(64, BB);

  // set-norm stats + normalize inputs
  k_stats_part<<<sgrid, 256, 0, stream>>>(query, part, perBatch);
  k_stats_final<<<BB, 32, 0, stream>>>(part, stats + 0);
  k_stats_part<<<sgrid, 256, 0, stream>>>(keyv, part, perBatch);
  k_stats_final<<<BB, 32, 0, stream>>>(part, stats + 4);
  k_norm<<<2048, 256, 0, stream>>>(query, stats + 0, gq, btq, qn, perBatch, invCount);
  k_norm<<<2048, 256, 0, stream>>>(keyv,  stats + 4, gk, btk, kvn, perBatch, invCount);

  // pack weights into WMMA B layout
  k_pack<<<64, 256, 0, stream>>>(Wq, Wp + 0 * 262144);
  k_pack<<<64, 256, 0, stream>>>(Wk, Wp + 1 * 262144);
  k_pack<<<64, 256, 0, stream>>>(Wv, Wp + 2 * 262144);
  k_pack<<<64, 256, 0, stream>>>(Wo, Wp + 3 * 262144);

  // projections (V stored transposed for attention B-operand)
  k_gemm<<<MROWS / 16, 256, 0, stream>>>(qn,  Wp + 0 * 262144, bq, Qm,  nullptr, nullptr, 0);
  k_gemm<<<MROWS / 16, 256, 0, stream>>>(kvn, Wp + 1 * 262144, bk, Km,  nullptr, nullptr, 0);
  k_gemm<<<MROWS / 16, 256, 0, stream>>>(kvn, Wp + 2 * 262144, bv, VTt, nullptr, nullptr, 1);

  // attention + residual into O1
  k_attn<<<(BB * HH * (NSEQ / 16)) / 8, 256, 0, stream>>>(Qm, Km, VTt, query, O1);

  // output set-norm + fused GEMM/ReLU/residual
  k_stats_part<<<sgrid, 256, 0, stream>>>(O1, part, perBatch);
  k_stats_final<<<BB, 32, 0, stream>>>(part, stats + 8);
  k_norm<<<2048, 256, 0, stream>>>(O1, stats + 8, go, bto, On, perBatch, invCount);
  k_gemm<<<MROWS / 16, 256, 0, stream>>>(On, Wp + 3 * 262144, bo, nullptr, O1, out, 2);
}
